// KNNComputerNoCheck_40604620817222
// MI455X (gfx1250) — matrix operand order repaired and checked by reading, not compiled, not run on hardware
//
#include <hip/hip_runtime.h>
#include <stdint.h>

typedef __attribute__((ext_vector_type(16))) __bf16 v16bf;
typedef __attribute__((ext_vector_type(8)))  float  v8f;
typedef unsigned short u16;
typedef __attribute__((ext_vector_type(4)))  u16 v4u;
typedef __attribute__((ext_vector_type(8)))  u16 v8u;
typedef __attribute__((ext_vector_type(16))) u16 v16u;
typedef __attribute__((ext_vector_type(4)))  int v4i;
typedef __attribute__((address_space(1))) v4i* gv4i_p;
typedef __attribute__((address_space(3))) v4i* lv4i_p;

#if defined(__has_builtin)
#if __has_builtin(__builtin_amdgcn_global_load_async_to_lds_b128) && \
    __has_builtin(__builtin_amdgcn_s_wait_asynccnt)
#define HAVE_ASYNC_LDS 1
#endif
#endif
#ifndef HAVE_ASYNC_LDS
#define HAVE_ASYNC_LDS 0
#endif

// round-to-nearest-even fp32 -> bf16 bits
__device__ __forceinline__ u16 f2bf_rne(float f) {
    unsigned u = __float_as_uint(f);
    return (u16)((u + 0x7FFFu + ((u >> 16) & 1u)) >> 16);
}

__device__ __forceinline__ void split1(float f, u16& h, u16& l) {
    u16 hb = f2bf_rne(f);
    h = hb;
    l = f2bf_rne(f - __uint_as_float((unsigned)hb << 16));
}

// 16-byte global -> LDS copy: async DMA path (ASYNCcnt-tracked) if the
// toolchain exposes it, else a synchronous VGPR relay (b128 load + ds_store).
__device__ __forceinline__ void copy16_g2lds(u16* lds_dst, const u16* gsrc) {
#if HAVE_ASYNC_LDS
    __builtin_amdgcn_global_load_async_to_lds_b128(
        (gv4i_p)(gsrc), (lv4i_p)(lds_dst), 0, 0);
#else
    *(v8u*)lds_dst = *(const v8u*)gsrc;
#endif
}

__device__ __forceinline__ void wait_async_all() {
#if HAVE_ASYNC_LDS
    __builtin_amdgcn_s_wait_asynccnt(0);
#endif
}

// ---- Kernel 1: fused row-norms + one-time fp32 -> bf16(hi)/bf16(lo) conversion ----
__global__ __launch_bounds__(256)
void knn_prep(const float* __restrict__ x, const float* __restrict__ y,
              float* __restrict__ xn, float* __restrict__ yn,
              u16* __restrict__ Xh, u16* __restrict__ Xl,
              u16* __restrict__ Yh, u16* __restrict__ Yl,
              unsigned long long* __restrict__ rowmin, int B, int D) {
    int row = blockIdx.x;
    bool isx = row < B;
    int r = isx ? row : row - B;
    const float4* p = (const float4*)((isx ? x : y) + (size_t)r * D);
    u16* Hr = (isx ? Xh : Yh) + (size_t)r * D;
    u16* Lr = (isx ? Xl : Yl) + (size_t)r * D;
    int n4 = D >> 2;
    float s = 0.f;
    for (int i = threadIdx.x; i < n4; i += 256) {
        float4 v = p[i];
        s += v.x * v.x + v.y * v.y + v.z * v.z + v.w * v.w;
        v4u h, l;
        u16 hh, ll;
        split1(v.x, hh, ll); h[0] = hh; l[0] = ll;
        split1(v.y, hh, ll); h[1] = hh; l[1] = ll;
        split1(v.z, hh, ll); h[2] = hh; l[2] = ll;
        split1(v.w, hh, ll); h[3] = hh; l[3] = ll;
        *(v4u*)(Hr + 4 * i) = h;
        *(v4u*)(Lr + 4 * i) = l;
    }
    __shared__ float red[256];
    red[threadIdx.x] = s;
    __syncthreads();
    for (int off = 128; off > 0; off >>= 1) {
        if (threadIdx.x < off) red[threadIdx.x] += red[threadIdx.x + off];
        __syncthreads();
    }
    if (threadIdx.x == 0) {
        if (isx) { xn[r] = red[0]; rowmin[r] = ~0ull; }
        else     { yn[r] = red[0]; }
    }
}

// ---- Kernel 2: bf16x3 WMMA GEMM + fused distance row-min ----
// Block = 256 threads = 8 waves; 64x64 output tile; double-buffered async LDS staging.
// Wave w: rows [(w>>1)*16,+16), cols [(w&1)*32,+32) (two 16x16 WMMA tiles).
__global__ __launch_bounds__(256)
void knn_gemm_rowmin(const u16* __restrict__ Xh, const u16* __restrict__ Xl,
                     const u16* __restrict__ Yh, const u16* __restrict__ Yl,
                     const float* __restrict__ xn, const float* __restrict__ yn,
                     unsigned long long* __restrict__ rowmin, int D) {
    __shared__ __align__(64) u16 sXh[2][64][32];
    __shared__ __align__(64) u16 sXl[2][64][32];
    __shared__ __align__(64) u16 sYh[2][64][32];
    __shared__ __align__(64) u16 sYl[2][64][32];

    const int tid  = threadIdx.x;
    const int lane = tid & 31;
    const int w    = tid >> 5;
    const int wm   = w >> 1;       // 0..3 : row quadrant within 64
    const int wn   = w & 1;        // 0..1 : 32-col half within 64
    const int kh   = lane >> 4;    // half-wave (K-half select in frags)
    const int ln   = lane & 15;    // M (A) / N (B,C,D)

    const int xbase = blockIdx.y * 64;
    const int ybase = blockIdx.x * 64;

    // copy mapping: 64x32 u16 tile = 256 x 16B; thread -> (row = tid/4, col = (tid%4)*8)
    const int lr = tid >> 2;
    const int lc = (tid & 3) << 3;
    const u16* gxh = Xh + (size_t)(xbase + lr) * D + lc;
    const u16* gxl = Xl + (size_t)(xbase + lr) * D + lc;
    const u16* gyh = Yh + (size_t)(ybase + lr) * D + lc;
    const u16* gyl = Yl + (size_t)(ybase + lr) * D + lc;

    v8f acc0 = {};
    v8f acc1 = {};

    // prologue: stage chunk 0 into buffer 0
    copy16_g2lds(&sXh[0][lr][lc], gxh);
    copy16_g2lds(&sXl[0][lr][lc], gxl);
    copy16_g2lds(&sYh[0][lr][lc], gyh);
    copy16_g2lds(&sYl[0][lr][lc], gyl);

    int buf = 0;
    for (int kc = 0; kc < D; kc += 32) {
        wait_async_all();     // our contribution to buffer `buf` landed in LDS
        __syncthreads();      // everyone's did; prior reads of buf^1 are done

        if (kc + 32 < D) {    // prefetch next chunk into the other buffer
            copy16_g2lds(&sXh[buf ^ 1][lr][lc], gxh + kc + 32);
            copy16_g2lds(&sXl[buf ^ 1][lr][lc], gxl + kc + 32);
            copy16_g2lds(&sYh[buf ^ 1][lr][lc], gyh + kc + 32);
            copy16_g2lds(&sYl[buf ^ 1][lr][lc], gyl + kc + 32);
        }

        // ---- A fragments (16x32 bf16, M = ln, contiguous 16-element K run) ----
        const u16* xrh = &sXh[buf][(wm << 4) + ln][0];
        const u16* xrl = &sXl[buf][(wm << 4) + ln][0];
        v8u ah_a = *(const v8u*)(xrh + (kh << 4));
        v8u ah_b = *(const v8u*)(xrh + 8 + (kh << 4));
        v8u al_a = *(const v8u*)(xrl + (kh << 4));
        v8u al_b = *(const v8u*)(xrl + 8 + (kh << 4));
        v16bf Ah = __builtin_bit_cast(v16bf, __builtin_shufflevector(
            ah_a, ah_b, 0, 1, 2, 3, 4, 5, 6, 7, 8, 9, 10, 11, 12, 13, 14, 15));
        v16bf Al = __builtin_bit_cast(v16bf, __builtin_shufflevector(
            al_a, al_b, 0, 1, 2, 3, 4, 5, 6, 7, 8, 9, 10, 11, 12, 13, 14, 15));

        // ---- B fragments (32x16 bf16, N = ln, contiguous 16-element K run) ----
        const int colA = (wn << 5) + ln;
        const int colB = colA + 16;
        v16bf Bh0 = __builtin_bit_cast(v16bf, *(const v16u*)(&sYh[buf][colA][kh << 4]));
        v16bf Bl0 = __builtin_bit_cast(v16bf, *(const v16u*)(&sYl[buf][colA][kh << 4]));
        v16bf Bh1 = __builtin_bit_cast(v16bf, *(const v16u*)(&sYh[buf][colB][kh << 4]));
        v16bf Bl1 = __builtin_bit_cast(v16bf, *(const v16u*)(&sYl[buf][colB][kh << 4]));

        // bf16x3: D += Ah*Bh + Ah*Bl + Al*Bh (Al*Bl term below fp32 noise floor)
        acc0 = __builtin_amdgcn_wmma_f32_16x16x32_bf16(false, Ah, false, Bh0, (short)0, acc0, false, false);
        acc0 = __builtin_amdgcn_wmma_f32_16x16x32_bf16(false, Ah, false, Bl0, (short)0, acc0, false, false);
        acc0 = __builtin_amdgcn_wmma_f32_16x16x32_bf16(false, Al, false, Bh0, (short)0, acc0, false, false);
        acc1 = __builtin_amdgcn_wmma_f32_16x16x32_bf16(false, Ah, false, Bh1, (short)0, acc1, false, false);
        acc1 = __builtin_amdgcn_wmma_f32_16x16x32_bf16(false, Ah, false, Bl1, (short)0, acc1, false, false);
        acc1 = __builtin_amdgcn_wmma_f32_16x16x32_bf16(false, Al, false, Bh1, (short)0, acc1, false, false);

        buf ^= 1;
    }

    // ---- epilogue: d2 = x2 + y2 - 2*dot, clamp, row-wise min+argmin ----
    const int colA = ybase + (wn << 5) + ln;
    const int colB = colA + 16;
    const float y2a = yn[colA];
    const float y2b = yn[colB];

    #pragma unroll
    for (int r = 0; r < 8; ++r) {
        const int grow = xbase + (wm << 4) + r + (kh << 3);  // C layout: VGPR r -> M=r / r+8
        const float x2v = xn[grow];
        float d2a = fmaxf(x2v + y2a - 2.f * acc0[r], 0.f);
        float d2b = fmaxf(x2v + y2b - 2.f * acc1[r], 0.f);
        float v = d2a; int c = colA;
        if (d2b < v) { v = d2b; c = colB; }
        #pragma unroll
        for (int off = 8; off > 0; off >>= 1) {
            float ov = __shfl_xor(v, off, 16);
            int   oc = __shfl_xor(c, off, 16);
            if (ov < v || (ov == v && oc < c)) { v = ov; c = oc; }
        }
        if (ln == 0) {
            unsigned long long key =
                ((unsigned long long)__float_as_uint(v) << 32) | (unsigned)c;
            atomicMin(&rowmin[grow], key);
        }
    }
}

// ---- Kernel 3: finalize scatter-update into d_out ----
__global__ __launch_bounds__(256)
void knn_finalize(const float* __restrict__ min_dists, const int* __restrict__ nn_indices,
                  const unsigned long long* __restrict__ rowmin,
                  const int* __restrict__ xs_p, const int* __restrict__ ys_p,
                  float* __restrict__ out, int N, int B) {
    int i = blockIdx.x * 256 + threadIdx.x;
    if (i >= N) return;
    int xs = xs_p[0];
    int ys = ys_p[0];
    float md = min_dists[i];
    float ni = (float)nn_indices[i];
    if (i >= xs && i < xs + B) {
        unsigned long long key = rowmin[i - xs];
        float d2   = __uint_as_float((unsigned)(key >> 32));
        md = fminf(sqrtf(d2), md);
        ni = (float)((int)(key & 0xFFFFFFFFull) + ys);
    }
    out[i]     = md;
    out[N + i] = ni;
}

extern "C" void kernel_launch(void* const* d_in, const int* in_sizes, int n_in,
                              void* d_out, int out_size, void* d_ws, size_t ws_size,
                              hipStream_t stream) {
    const float* x  = (const float*)d_in[0];
    const float* y  = (const float*)d_in[1];
    const float* md = (const float*)d_in[2];
    const int*   nn = (const int*)d_in[3];
    const int*   xs = (const int*)d_in[4];
    const int*   ys = (const int*)d_in[5];

    const int D = 3072;
    const int B = in_sizes[0] / D;     // 4096
    const int N = in_sizes[2];         // 50000

    // Workspace layout (assumes ws_size >= ~101 MB):
    //   [xn: B f32][yn: B f32][rowmin: B u64][Xh,Xl,Yh,Yl: B*D u16 each]
    char* wsb = (char*)d_ws;
    float* xn = (float*)wsb;
    float* yn = xn + B;
    unsigned long long* rowmin = (unsigned long long*)(wsb + (size_t)2 * B * sizeof(float));
    u16* Xh = (u16*)(wsb + (size_t)2 * B * sizeof(float) + (size_t)B * 8);
    u16* Xl = Xh + (size_t)B * D;
    u16* Yh = Xl + (size_t)B * D;
    u16* Yl = Yh + (size_t)B * D;

    knn_prep<<<2 * B, 256, 0, stream>>>(x, y, xn, yn, Xh, Xl, Yh, Yl, rowmin, B, D);

    dim3 grid(B / 64, B / 64);
    knn_gemm_rowmin<<<grid, 256, 0, stream>>>(Xh, Xl, Yh, Yl, xn, yn, rowmin, D);

    knn_finalize<<<(N + 255) / 256, 256, 0, stream>>>(
        md, nn, rowmin, xs, ys, (float*)d_out, N, B);
}
